// CaptionNet_78030965833991
// MI455X (gfx1250) — matrix operand order repaired and checked by reading, not compiled, not run on hardware
//
#include <hip/hip_runtime.h>
#include <hip/hip_bf16.h>
#include <math.h>

// ---------------------------------------------------------------------------
// Problem constants (from reference)
// ---------------------------------------------------------------------------
#define BB 64
#define TT 20
#define CC 512
#define HW 196
#define VV 10000
#define EE 256
#define UU 512
#define KDIM 1280           // C+E+U = xh / X / W_out inner dim
#define G4U  2048           // 4*U

typedef __attribute__((ext_vector_type(16))) __bf16 v16bf;
typedef __attribute__((ext_vector_type(8)))  float  v8f;
typedef __attribute__((ext_vector_type(4)))  unsigned int u32x4;
typedef __attribute__((ext_vector_type(8)))  int i32x8;
typedef __attribute__((ext_vector_type(4)))  int i32x4;

// ---------------------------------------------------------------------------
// WMMA fragment loaders (gfx1250 wave32 layouts, CDNA5 ISA §7.12.2)
// A: 16x32 bf16.  lanes 0-15: row M=lane, K = {0..7, 16..23}
//                 lanes 16-31: row M=lane-16, K = {8..15, 24..31}
// ---------------------------------------------------------------------------
__device__ inline v16bf load_frag_a(const __bf16* __restrict__ A, int lda,
                                    int m0, int k0, int lane) {
  const int l  = lane & 15;
  const int hi = lane >> 4;
  const __bf16* p = A + (size_t)(m0 + l) * lda + k0 + hi * 8;
  union { uint4 q[2]; v16bf v; } u;
  u.q[0] = *(const uint4*)(p);        // 8 bf16: K = k0+hi*8 .. +7
  u.q[1] = *(const uint4*)(p + 16);   // 8 bf16: K = k0+16+hi*8 .. +7
  return u.v;
}

// B: 32x16 bf16 (B source stored as W[n][k], row-major over K).
// lane n (0-15): column N=n, K = {0..7, 16..23} ; lane n+16: column N=n, K = {8..15, 24..31}
__device__ inline v16bf load_frag_b(const __bf16* __restrict__ W, int ldb,
                                    int n0, int k0, int lane) {
  const int l  = lane & 15;
  const int hi = lane >> 4;
  const __bf16* p = W + (size_t)(n0 + l) * ldb + k0 + hi * 8;
  union { uint4 q[2]; v16bf v; } u;
  u.q[0] = *(const uint4*)(p);        // K = k0+hi*8 .. +7
  u.q[1] = *(const uint4*)(p + 16);   // K = k0+16+hi*8 .. +7
  return u.v;
}

// ---------------------------------------------------------------------------
// Generic wave-tiled bf16 WMMA GEMM, ping-pong double-buffered (K % 64 == 0):
//   D[M,N] = A[M,K] * Bw[N,K]^T + bias
// One wave computes a 16 x (16*NT) output tile.  nGroups = N/(16*NT).
// Two fragment buffer sets alternate producer/consumer roles so the compiler
// needs no rotation copies; loads of slab k+32 overlap the WMMAs of slab k.
// ---------------------------------------------------------------------------
template <int NT>
__global__ void k_wmma_gemm(const __bf16* __restrict__ A, int lda,
                            const __bf16* __restrict__ Bw, int ldb,
                            float* __restrict__ D, int ldd,
                            const float* __restrict__ bias,
                            int K, int nGroups, int nWaves) {
  const int wave = (blockIdx.x * blockDim.x + threadIdx.x) >> 5;
  if (wave >= nWaves) return;              // wave-uniform: EXEC stays all-1s
  const int lane = threadIdx.x & 31;
  const int mT = wave / nGroups;
  const int nG = wave - mT * nGroups;
  const int m0 = mT * 16;
  const int n0 = nG * (16 * NT);
  const int l = lane & 15;

  v8f acc[NT];
#pragma unroll
  for (int j = 0; j < NT; ++j) acc[j] = (v8f)(0.0f);

  // buffer set 0 holds slab k0, set 1 holds slab k0+32
  v16bf aA = load_frag_a(A, lda, m0, 0, lane);
  v16bf bA[NT];
#pragma unroll
  for (int j = 0; j < NT; ++j) bA[j] = load_frag_b(Bw, ldb, n0 + 16 * j, 0, lane);

  for (int k0 = 0; k0 < K; k0 += 64) {     // requires K % 64 == 0
    // issue loads for slab k0+32, then consume slab k0
    const int k1 = k0 + 32;
    __builtin_prefetch(A + (size_t)(m0 + l) * lda + k1 + 32, 0, 1);
    v16bf aB = load_frag_a(A, lda, m0, k1, lane);
    v16bf bB[NT];
#pragma unroll
    for (int j = 0; j < NT; ++j) bB[j] = load_frag_b(Bw, ldb, n0 + 16 * j, k1, lane);
#pragma unroll
    for (int j = 0; j < NT; ++j) {
      acc[j] = __builtin_amdgcn_wmma_f32_16x16x32_bf16(
          false, aA, false, bA[j], (short)0, acc[j], false, false);
    }
    // issue loads for slab k0+64 (if any), then consume slab k0+32
    const int k2 = k0 + 64;
    if (k2 < K) {
      __builtin_prefetch(A + (size_t)(m0 + l) * lda + k2 + 32, 0, 1);
      aA = load_frag_a(A, lda, m0, k2, lane);
#pragma unroll
      for (int j = 0; j < NT; ++j)
        bA[j] = load_frag_b(Bw, ldb, n0 + 16 * j, k2, lane);
    }
#pragma unroll
    for (int j = 0; j < NT; ++j) {
      acc[j] = __builtin_amdgcn_wmma_f32_16x16x32_bf16(
          false, aB, false, bB[j], (short)0, acc[j], false, false);
    }
  }

  const int hi = lane >> 4;
#pragma unroll
  for (int j = 0; j < NT; ++j) {
    const int col = n0 + 16 * j + l;
    const float bv = bias ? bias[col] : 0.0f;
#pragma unroll
    for (int r = 0; r < 8; ++r) {
      const int row = m0 + hi * 8 + r;     // C/D layout: VGPR r -> M=r / M=r+8
      D[(size_t)row * ldd + col] = acc[j][r] + bv;
    }
  }
}

// ---------------------------------------------------------------------------
// keys GEMM with TDM-staged A tile:
//   block = one 16-row M tile; TDM loads the full 16x512 bf16 A slab (16 KB)
//   into LDS once; 8 waves each produce 64 of the 512 output columns, reading
//   A fragments from LDS (ds_load_b128) and W_key fragments from global.
// `fill_fallback` (host passes 0) is a runtime-opaque guard whose dead store
// path makes the LDS tile visibly written, so the compiler cannot fold the
// ds reads to undef (the TDM builtin is opaque to alias analysis).
// ---------------------------------------------------------------------------
__global__ void __launch_bounds__(256)
k_keys_wmma_lds(const __bf16* __restrict__ A,    // feats bf16 [12544, 512]
                const __bf16* __restrict__ Bw,   // W_key bf16 [512, 512]
                float* __restrict__ D,           // keys f32  [12544, 512]
                const float* __restrict__ bias,
                int fill_fallback) {
  __shared__ __bf16 atile[16 * CC];              // 16 KB, LDS offset 0
  const int m0 = blockIdx.x * 16;
  const int lane = threadIdx.x & 31;
  const int wave = threadIdx.x >> 5;             // 0..7

  if (wave == 0) {
    // Tensor DMA descriptor (CDNA5 ISA):
    // 2D tile: tile_dim0=512 elems/row, tile_dim1=16 rows, data_size=2B,
    // tensor_dim0_stride=512 (row-major A), dest LDS addr 0, D#.type=2.
    unsigned long long ga = (unsigned long long)(uintptr_t)(A + (size_t)m0 * CC);
    u32x4 g0;
    g0[0] = 1u;                                          // count=1, no gather
    g0[1] = 0u;                                          // lds_addr = 0
    g0[2] = (unsigned)(ga & 0xffffffffull);              // global_addr[31:0]
    g0[3] = (unsigned)((ga >> 32) & 0x1ffffffull)        // global_addr[56:32]
            | (2u << 30);                                // type=2 ("image")
    i32x8 g1;
    g1[0] = (int)(1u << 16);                             // data_size=1 -> 2B
    g1[1] = (int)(((unsigned)CC & 0xffffu) << 16);       // tensor_dim0[15:0]
    g1[2] = (int)(16u << 16);                            // dim0[31:16]=0 | tensor_dim1=16
    g1[3] = (int)((unsigned)CC << 16);                   // dim1[31:16]=0 | tile_dim0=512
    g1[4] = 16;                                          // tile_dim1=16 | tile_dim2=0
    g1[5] = CC;                                          // tensor_dim0_stride[31:0]
    g1[6] = 0;                                           // stride0 hi | stride1 lo
    g1[7] = 0;                                           // stride1 hi
    i32x4 gz = (i32x4)0;                                 // groups 2/3 unused (2D)
#if defined(__clang_major__) && (__clang_major__ >= 23)
    i32x8 gz8 = (i32x8)0;
    __builtin_amdgcn_tensor_load_to_lds(g0, g1, gz, gz, gz8, 0);
#else
    __builtin_amdgcn_tensor_load_to_lds(g0, g1, gz, gz, 0);
#endif
    __builtin_amdgcn_s_wait_tensorcnt(0);
  }
  if (fill_fallback) {
    // Dead at runtime (host passes 0) but keeps atile visibly written.
    for (int i = threadIdx.x; i < 16 * CC; i += 256)
      atile[i] = A[(size_t)m0 * CC + i];
  }
  __syncthreads();

  const int n0 = wave * 64;                      // 4 N-tiles per wave
  v8f acc[4];
#pragma unroll
  for (int j = 0; j < 4; ++j) acc[j] = (v8f)(0.0f);

  // ping-pong pipeline: LDS A frag + global B frags of slab k+32 overlap WMMAs
  v16bf aA = load_frag_a(atile, CC, 0, 0, lane);         // ds_load_b128 path
  v16bf bA[4];
#pragma unroll
  for (int j = 0; j < 4; ++j) bA[j] = load_frag_b(Bw, CC, n0 + 16 * j, 0, lane);

  for (int k0 = 0; k0 < CC; k0 += 64) {          // CC % 64 == 0
    const int k1 = k0 + 32;
    v16bf aB = load_frag_a(atile, CC, 0, k1, lane);
    v16bf bB[4];
#pragma unroll
    for (int j = 0; j < 4; ++j) bB[j] = load_frag_b(Bw, CC, n0 + 16 * j, k1, lane);
#pragma unroll
    for (int j = 0; j < 4; ++j) {
      acc[j] = __builtin_amdgcn_wmma_f32_16x16x32_bf16(
          false, aA, false, bA[j], (short)0, acc[j], false, false);
    }
    const int k2 = k0 + 64;
    if (k2 < CC) {
      aA = load_frag_a(atile, CC, 0, k2, lane);
#pragma unroll
      for (int j = 0; j < 4; ++j)
        bA[j] = load_frag_b(Bw, CC, n0 + 16 * j, k2, lane);
    }
#pragma unroll
    for (int j = 0; j < 4; ++j) {
      acc[j] = __builtin_amdgcn_wmma_f32_16x16x32_bf16(
          false, aB, false, bB[j], (short)0, acc[j], false, false);
    }
  }

  const int l = lane & 15;
  const int hi = lane >> 4;
#pragma unroll
  for (int j = 0; j < 4; ++j) {
    const int col = n0 + 16 * j + l;
    const float bv = bias[col];
#pragma unroll
    for (int r = 0; r < 8; ++r) {
      const int row = m0 + hi * 8 + r;
      D[(size_t)row * UU + col] = acc[j][r] + bv;
    }
  }
}

// ---------------------------------------------------------------------------
// Prep kernels
// ---------------------------------------------------------------------------
__global__ void k_feats_mean(const float* __restrict__ in, float* __restrict__ out) {
  int idx = blockIdx.x * blockDim.x + threadIdx.x;   // b*C + c
  if (idx >= BB * CC) return;
  const float* p = in + (size_t)idx * HW;
  float s = 0.f;
  for (int h = 0; h < HW; ++h) s += p[h];
  out[idx] = s * (1.0f / (float)HW);
}

// feats_bf16[b][hw][c] = image_features[b][c][hw]  (transpose + bf16)
__global__ void k_feats_bf16(const float* __restrict__ in, __bf16* __restrict__ out) {
  size_t idx = (size_t)blockIdx.x * blockDim.x + threadIdx.x;
  if (idx >= (size_t)BB * HW * CC) return;
  size_t c = idx % CC;
  size_t rem = idx / CC;
  size_t hw = rem % HW;
  size_t b = rem / HW;
  out[idx] = (__bf16)in[(b * CC + c) * HW + hw];
}

__global__ void k_f32_to_bf16(const float* __restrict__ in, __bf16* __restrict__ out,
                              size_t n) {
  size_t idx = (size_t)blockIdx.x * blockDim.x + threadIdx.x;
  if (idx < n) out[idx] = (__bf16)in[idx];
}

// W_cat[n][k] : k<768 -> W_ih[n][k] (x = [emb, a]); k>=768 -> W_hh[n][k-768]
__global__ void k_make_wcat(const float* __restrict__ Wih, const float* __restrict__ Whh,
                            __bf16* __restrict__ out) {
  size_t idx = (size_t)blockIdx.x * blockDim.x + threadIdx.x;
  if (idx >= (size_t)G4U * KDIM) return;
  size_t n = idx / KDIM;
  size_t k = idx % KDIM;
  float v = (k < (CC + EE)) ? Wih[n * (CC + EE) + k] : Whh[n * UU + (k - (CC + EE))];
  out[idx] = (__bf16)v;
}

__global__ void k_bias_sum(const float* __restrict__ bih, const float* __restrict__ bhh,
                           float* __restrict__ out) {
  int i = blockIdx.x * blockDim.x + threadIdx.x;
  if (i < G4U) out[i] = bih[i] + bhh[i];
}

// hid0/cell0: small f32 GEMV per output element (B*U = 32768 dots over C)
__global__ void k_init_state(const float* __restrict__ fm,
                             const float* __restrict__ Wh, const float* __restrict__ bh,
                             const float* __restrict__ Wc, const float* __restrict__ bc,
                             float* __restrict__ hid, float* __restrict__ cell) {
  int idx = blockIdx.x * blockDim.x + threadIdx.x;
  if (idx >= BB * UU) return;
  int b = idx >> 9, u = idx & (UU - 1);
  const float* f  = fm + (size_t)b * CC;
  const float* wh = Wh + (size_t)u * CC;
  const float* wc = Wc + (size_t)u * CC;
  float sh = 0.f, sc = 0.f;
  for (int c = 0; c < CC; ++c) { float fv = f[c]; sh += fv * wh[c]; sc += fv * wc[c]; }
  hid[idx]  = sh + bh[u];
  cell[idx] = sc + bc[u];
}

// Embedding gather with max_norm=5 renorm. One block (256 threads == E) per row.
__global__ void k_embed(const float* __restrict__ emb, const int* __restrict__ ix,
                        float* __restrict__ cap_emb, __bf16* __restrict__ X) {
  int r = blockIdx.x;            // r = b*T + t
  int tid = threadIdx.x;         // 0..255 == e
  __shared__ float red[EE];
  int tok = ix[r];
  float v = emb[(size_t)tok * EE + tid];
  red[tid] = v * v;
  __syncthreads();
  for (int off = EE / 2; off > 0; off >>= 1) {
    if (tid < off) red[tid] += red[tid + off];
    __syncthreads();
  }
  float nrm = sqrtf(red[0]);
  float scale = fminf(1.0f, 5.0f / fmaxf(nrm, 1e-12f));
  float o = v * scale;
  cap_emb[(size_t)r * EE + tid] = o;
  X[(size_t)r * KDIM + (UU + CC) + tid] = (__bf16)o;   // emb_t slot of out row
}

// ---------------------------------------------------------------------------
// Per-step attention: scores -> softmax -> context. One block per batch elem.
// ---------------------------------------------------------------------------
__global__ void k_attention(const float* __restrict__ keys, const float* __restrict__ hid,
                            const float* __restrict__ imgfeat,
                            const float* __restrict__ cap_emb,
                            __bf16* __restrict__ xh, __bf16* __restrict__ X,
                            float* __restrict__ attn_out, int t) {
  const int b = blockIdx.x;
  const int tid = threadIdx.x;
  __shared__ float w[224];       // softmax weights (padded past HW=196)
  __shared__ float red[256];
  const size_t r = (size_t)b * TT + t;

  // scores[k] = (hid[b] . keys[b][k]) / sqrt(U)
  float sc = -1e30f;
  if (tid < HW) {
    const float* kk = keys + ((size_t)b * HW + tid) * UU;
    const float* hh = hid + (size_t)b * UU;
    float s = 0.f;
    for (int u = 0; u < UU; ++u) s += hh[u] * kk[u];
    sc = s * 0.044194173824159216f;   // 1/sqrt(512)
  }
  red[tid] = sc;
  __syncthreads();
  for (int off = 128; off > 0; off >>= 1) {
    if (tid < off) red[tid] = fmaxf(red[tid], red[tid + off]);
    __syncthreads();
  }
  const float mx = red[0];
  __syncthreads();
  float e = 0.f;
  if (tid < HW) e = expf(sc - mx);
  red[tid] = e;
  __syncthreads();
  for (int off = 128; off > 0; off >>= 1) {
    if (tid < off) red[tid] += red[tid + off];
    __syncthreads();
  }
  const float inv = 1.0f / red[0];
  __syncthreads();
  if (tid < HW) {
    float wv = e * inv;
    w[tid] = wv;
    attn_out[r * HW + tid] = wv;               // attn output [B,T,HW]
  }
  __syncthreads();

  // a[c] = sum_k w[k] * image_features[b][c][k]
  for (int c = tid; c < CC; c += blockDim.x) {
    const float* f = imgfeat + ((size_t)b * CC + c) * HW;
    float s = 0.f;
    for (int k = 0; k < HW; ++k) s += w[k] * f[k];
    __bf16 bv = (__bf16)s;
    xh[(size_t)b * KDIM + EE + c] = bv;        // xh a slot (x = [emb, a])
    X[r * KDIM + UU + c] = bv;                 // out row a slot ([hid, a, emb])
  }
  // xh emb slot
  if (tid < EE) xh[(size_t)b * KDIM + tid] = (__bf16)cap_emb[r * EE + tid];
  // xh hid slot (pre-update hidden state)
  for (int u = tid; u < UU; u += blockDim.x)
    xh[(size_t)b * KDIM + EE + CC + u] = (__bf16)hid[(size_t)b * UU + u];
}

// LSTM pointwise update; writes new hid/cell and X[:, 0..U] = hid_new (bf16)
__global__ void k_lstm(const float* __restrict__ gates,
                       float* __restrict__ hid, float* __restrict__ cell,
                       __bf16* __restrict__ X, int t) {
  int idx = blockIdx.x * blockDim.x + threadIdx.x;
  if (idx >= BB * UU) return;
  int b = idx >> 9, u = idx & (UU - 1);
  const float* g = gates + (size_t)b * G4U;
  float gi = g[u], gf = g[UU + u], gg = g[2 * UU + u], go = g[3 * UU + u];
  float si = 1.0f / (1.0f + expf(-gi));
  float sf = 1.0f / (1.0f + expf(-gf));
  float so = 1.0f / (1.0f + expf(-go));
  float c = sf * cell[idx] + si * tanhf(gg);
  float h = so * tanhf(c);
  cell[idx] = c;
  hid[idx] = h;
  X[((size_t)b * TT + t) * KDIM + u] = (__bf16)h;
}

// ---------------------------------------------------------------------------
// Host-side orchestration
// ---------------------------------------------------------------------------
static inline size_t align256(size_t x) { return (x + 255) & ~(size_t)255; }

extern "C" void kernel_launch(void* const* d_in, const int* in_sizes, int n_in,
                              void* d_out, int out_size, void* d_ws, size_t ws_size,
                              hipStream_t stream) {
  (void)in_sizes; (void)n_in; (void)out_size; (void)ws_size;
  const float* imgfeat = (const float*)d_in[0];
  const int*   capix   = (const int*)  d_in[1];
  const float* W_h0    = (const float*)d_in[2];
  const float* b_h0    = (const float*)d_in[3];
  const float* W_c0    = (const float*)d_in[4];
  const float* b_c0    = (const float*)d_in[5];
  const float* emb     = (const float*)d_in[6];
  const float* W_key   = (const float*)d_in[7];
  const float* b_key   = (const float*)d_in[8];
  const float* W_ih    = (const float*)d_in[9];
  const float* b_ih    = (const float*)d_in[10];
  const float* W_hh    = (const float*)d_in[11];
  const float* b_hh    = (const float*)d_in[12];
  const float* W_out   = (const float*)d_in[13];
  const float* b_out   = (const float*)d_in[14];

  float* logits = (float*)d_out;                               // [B,T,V]
  float* attn   = (float*)d_out + (size_t)BB * TT * VV;        // [B,T,HW]

  // workspace carve-out
  char* ws = (char*)d_ws;
  size_t off = 0;
  auto carve = [&](size_t bytes) -> char* {
    char* p = ws + off;
    off = align256(off + bytes);
    return p;
  };
  float*  feats_mean = (float*) carve((size_t)BB * CC * 4);
  float*  hid        = (float*) carve((size_t)BB * UU * 4);
  float*  cell       = (float*) carve((size_t)BB * UU * 4);
  float*  cap_emb    = (float*) carve((size_t)BB * TT * EE * 4);
  float*  gates      = (float*) carve((size_t)BB * G4U * 4);
  float*  bsum       = (float*) carve((size_t)G4U * 4);
  float*  keys       = (float*) carve((size_t)BB * HW * UU * 4);
  __bf16* xh         = (__bf16*)carve((size_t)BB * KDIM * 2);
  __bf16* X          = (__bf16*)carve((size_t)BB * TT * KDIM * 2);
  __bf16* featsb     = (__bf16*)carve((size_t)BB * HW * CC * 2);
  __bf16* wkeyb      = (__bf16*)carve((size_t)UU * CC * 2);
  __bf16* wcatb      = (__bf16*)carve((size_t)G4U * KDIM * 2);
  __bf16* woutb      = (__bf16*)carve((size_t)VV * KDIM * 2);

  // ---- prep ----
  k_feats_mean<<<(BB * CC + 255) / 256, 256, 0, stream>>>(imgfeat, feats_mean);
  {
    size_t n = (size_t)BB * HW * CC;
    k_feats_bf16<<<(unsigned)((n + 255) / 256), 256, 0, stream>>>(imgfeat, featsb);
  }
  k_f32_to_bf16<<<(unsigned)(((size_t)UU * CC + 255) / 256), 256, 0, stream>>>(
      W_key, wkeyb, (size_t)UU * CC);
  k_f32_to_bf16<<<(unsigned)(((size_t)VV * KDIM + 255) / 256), 256, 0, stream>>>(
      W_out, woutb, (size_t)VV * KDIM);
  k_make_wcat<<<(unsigned)(((size_t)G4U * KDIM + 255) / 256), 256, 0, stream>>>(
      W_ih, W_hh, wcatb);
  k_bias_sum<<<(G4U + 255) / 256, 256, 0, stream>>>(b_ih, b_hh, bsum);
  k_init_state<<<(BB * UU + 255) / 256, 256, 0, stream>>>(
      feats_mean, W_h0, b_h0, W_c0, b_c0, hid, cell);
  k_embed<<<BB * TT, EE, 0, stream>>>(emb, capix, cap_emb, X);

  // ---- keys = feats_bf16 [12544,512] @ W_key^T  (TDM-staged A + WMMA) ----
  k_keys_wmma_lds<<<(BB * HW) / 16, 256, 0, stream>>>(featsb, wkeyb, keys, b_key,
                                                      /*fill_fallback=*/0);

  // ---- recurrent steps ----
  for (int t = 0; t < TT; ++t) {
    k_attention<<<BB, 256, 0, stream>>>(keys, hid, imgfeat, cap_emb, xh, X, attn, t);
    {
      const int mT = BB / 16;            // 4
      const int nGroups = G4U / 32;      // 64  (2 tiles/wave -> 256 waves)
      const int nWaves = mT * nGroups;   // 256
      k_wmma_gemm<2><<<nWaves / 8, 256, 0, stream>>>(
          xh, KDIM, wcatb, KDIM, gates, G4U, bsum, KDIM, nGroups, nWaves);
    }
    k_lstm<<<(BB * UU + 255) / 256, 256, 0, stream>>>(gates, hid, cell, X, t);
  }

  // ---- logits = X [1280,1280] @ W_out^T [10000,1280]  (WMMA, 5 tiles/wave) ----
  {
    const int mT = (BB * TT) / 16;       // 80
    const int nGroups = VV / 80;         // 125  (5 tiles/wave)
    const int nWaves = mT * nGroups;     // 10000
    k_wmma_gemm<5><<<nWaves / 8, 256, 0, stream>>>(
        X, KDIM, woutb, KDIM, logits, VV, b_out, KDIM, nGroups, nWaves);
  }
}
